// Modulated_Conv2d_24094766531363
// MI455X (gfx1250) — matrix operand order repaired and checked by reading, not compiled, not run on hardware
//
#include <hip/hip_runtime.h>
#include <stdint.h>

typedef __attribute__((ext_vector_type(16))) __bf16 v16bf;
typedef __attribute__((ext_vector_type(8)))  float  v8f;

#define BATCH 16
#define CIN   32
#define HH    512
#define WW    512
#define OCH   32
#define RTOT  288      // CIN*9, r = (kw*3+kh)*32 + c
#define WPITCH 292     // bf16 pitch for weight rows in LDS (146 dwords, conflict-free)
#define XPITCH 36      // bf16 per (row,col) cell: 32 channels + pad (18 dwords, conflict-free)
#define XCOLS 130      // 128-wide chunk + 2 halo
#define CHUNK 128

__device__ __forceinline__ unsigned short f2bf(float f) {
    union { float f; uint32_t u; } x; x.f = f;
    uint32_t u = x.u;
    uint32_t r = u + 0x7FFFu + ((u >> 16) & 1u);   // round-to-nearest-even
    return (unsigned short)(r >> 16);
}

// ---------------------------------------------------------------------------
// Kernel 1: style FC + LeakyReLU + modulate + demodulate -> bf16 weights
// layout wt_bf[b][oc][r], r = (kw*3+kh)*32 + c
// ---------------------------------------------------------------------------
__global__ void __launch_bounds__(256)
modw_kernel(const float* __restrict__ w,
            const float* __restrict__ weight,
            const float* __restrict__ style_scale,
            const float* __restrict__ fc_weight,
            const float* __restrict__ fc_bias,
            unsigned short* __restrict__ wt_bf)
{
    __shared__ float s_mod[CIN];
    __shared__ float wtmp[OCH][RTOT];
    __shared__ float psum[OCH][8];
    __shared__ float siginv[OCH];

    const int b = blockIdx.x;
    const int t = threadIdx.x;
    const float fc_scale = 0.04419417382415922f;   // 1/sqrt(512)
    const float w_scale  = 0.05892556509887896f;   // 1/sqrt(288)

    if (t < CIN) {
        float acc = 0.f;
        const float* wb = w + (size_t)b * 512;
        const float* fw = fc_weight + (size_t)t * 512;
        for (int k = 0; k < 512; ++k) acc += wb[k] * fw[k];
        float sv = acc * fc_scale + fc_bias[t];
        sv = (sv > 0.f) ? sv : 0.2f * sv;           // LeakyReLU(0.2)
        s_mod[t] = sv * (style_scale[0] + 1.0f);
    }
    __syncthreads();

    const int oc  = t >> 3;     // 8 threads per output channel
    const int sub = t & 7;
    float ss = 0.f;
    for (int rr = sub; rr < RTOT; rr += 8) {
        int c  = rr & 31;
        int g  = rr >> 5;       // 0..8 = kw*3+kh
        int kh = g % 3, kw = g / 3;
        float v = weight[(((size_t)oc * CIN + c) * 3 + kh) * 3 + kw] * w_scale * s_mod[c];
        wtmp[oc][rr] = v;
        ss += v * v;
    }
    psum[oc][sub] = ss;
    __syncthreads();
    if (sub == 0) {
        float tot = 0.f;
        for (int i = 0; i < 8; ++i) tot += psum[oc][i];
        siginv[oc] = rsqrtf(tot + 1e-8f);
    }
    __syncthreads();
    for (int idx = t; idx < OCH * RTOT; idx += 256) {
        int o = idx / RTOT, rr = idx - o * RTOT;
        wt_bf[((size_t)b * OCH + o) * RTOT + rr] = f2bf(wtmp[o][rr] * siginv[o]);
    }
}

// ---------------------------------------------------------------------------
// Kernel 2: implicit-GEMM grouped conv via v_wmma_f32_16x16x32_bf16
// block = (one output row h, one batch b); 8 waves x 16 cols = 128-col chunks
// ---------------------------------------------------------------------------
__global__ void __launch_bounds__(256, 2)
modconv_kernel(const float* __restrict__ x,
               const unsigned short* __restrict__ wt_bf,
               float* __restrict__ out)
{
    __shared__ unsigned short wt_lds[OCH * WPITCH];      // [oc][r], pitch 292 bf16
    __shared__ unsigned short xs[3 * XCOLS * XPITCH];    // [row][col][c], pitch 36 bf16

    const int b    = blockIdx.y;
    const int h    = blockIdx.x;
    const int t    = threadIdx.x;
    const int lane = t & 31;
    const int wv   = t >> 5;        // wave id 0..7
    const int hf   = lane >> 4;     // half-wave
    const int ln   = lane & 15;

    // stage per-batch bf16 weights into LDS (coalesced)
    for (int idx = t; idx < OCH * RTOT; idx += 256) {
        int o = idx / RTOT, rr = idx - o * RTOT;
        wt_lds[o * WPITCH + rr] = wt_bf[((size_t)b * OCH + o) * RTOT + rr];
    }

    const uint32_t* wt32 = (const uint32_t*)wt_lds;      // row pitch 146 dwords
    const uint32_t* xs32 = (const uint32_t*)xs;          // cell pitch 18 dwords
    const float* xb = x + (size_t)b * CIN * HH * WW;

    for (int chunk = 0; chunk < 4; ++chunk) {
        const int w0 = chunk * CHUNK;
        __syncthreads();    // weights ready (iter 0) / previous compute done

        // stage x halo tile -> bf16 LDS, layout [row][col][c]
        for (int e = t; e < CIN * 3 * XCOLS; e += 256) {
            int c   = e / (3 * XCOLS);
            int rem = e - c * 3 * XCOLS;
            int row = rem / XCOLS;
            int col = rem - row * XCOLS;
            int gh = h - 1 + row;
            int gw = w0 - 1 + col;
            float v = 0.f;
            if ((unsigned)gh < HH && (unsigned)gw < WW) {
                const float* p = &xb[(size_t)c * HH * WW + (size_t)gh * WW + gw];
                v = *p;
                if (chunk < 3 && ((col & 31) == 0) && gw + CHUNK < WW)
                    __builtin_prefetch(p + CHUNK, 0, 3);   // global_prefetch_b8: next chunk
            }
            xs[(row * XCOLS + col) * XPITCH + c] = f2bf(v);
        }
        __syncthreads();

        v8f acc0 = {0.f,0.f,0.f,0.f,0.f,0.f,0.f,0.f};
        v8f acc1 = {0.f,0.f,0.f,0.f,0.f,0.f,0.f,0.f};
        const int nout = wv * 16 + ln;      // output column within chunk

        #pragma unroll
        for (int s = 0; s < 9; ++s) {       // K step: 32 channels of tap (kh,kw)
            const int kh = s % 3, kw = s / 3;

            // B fragment (32x16 bf16): lane = col nout, half-wave splits c 0-15/16-31
            union { uint32_t u[8]; v16bf v; } bf;
            const uint32_t* bp = xs32 + (size_t)(kh * XCOLS + (nout + kw)) * (XPITCH / 2) + hf * 8;
            #pragma unroll
            for (int j = 0; j < 8; ++j) bf.u[j] = bp[j];

            // A fragments (16x32 bf16): row = oc (ln / ln+16), contiguous K (k = 16*hf + e), same as B
            union { uint32_t u[8]; v16bf v; } a0, a1;
            const int kbase = s * 16 + hf * 8;              // dword offset in row
            const uint32_t* ap0 = wt32 + (size_t)ln * (WPITCH / 2);
            const uint32_t* ap1 = wt32 + (size_t)(ln + 16) * (WPITCH / 2);
            #pragma unroll
            for (int j = 0; j < 4; ++j) {
                a0.u[j]     = ap0[kbase + j];
                a0.u[4 + j] = ap0[kbase + 4 + j];
                a1.u[j]     = ap1[kbase + j];
                a1.u[4 + j] = ap1[kbase + 4 + j];
            }

            acc0 = __builtin_amdgcn_wmma_f32_16x16x32_bf16(false, a0.v, false, bf.v,
                                                           (short)0, acc0, false, false);
            acc1 = __builtin_amdgcn_wmma_f32_16x16x32_bf16(false, a1.v, false, bf.v,
                                                           (short)0, acc1, false, false);
        }

        // D layout: lanes 0-15 VGPR v -> (M=v, N=ln); lanes 16-31 -> (M=v+8, N=ln)
        #pragma unroll
        for (int v = 0; v < 8; ++v) {
            const int oc0 = v + 8 * hf;
            const size_t col = (size_t)w0 + wv * 16 + ln;
            out[(((size_t)b * OCH + oc0) * HH + h) * WW + col]        = acc0[v];
            out[(((size_t)b * OCH + oc0 + 16) * HH + h) * WW + col]   = acc1[v];
        }
    }
}

extern "C" void kernel_launch(void* const* d_in, const int* in_sizes, int n_in,
                              void* d_out, int out_size, void* d_ws, size_t ws_size,
                              hipStream_t stream) {
    const float* x           = (const float*)d_in[0];
    const float* w           = (const float*)d_in[1];
    const float* weight      = (const float*)d_in[2];
    const float* style_scale = (const float*)d_in[3];
    const float* fc_weight   = (const float*)d_in[4];
    const float* fc_bias     = (const float*)d_in[5];
    unsigned short* wt_bf    = (unsigned short*)d_ws;   // 16*32*288*2 = 288 KB

    modw_kernel<<<dim3(BATCH), 256, 0, stream>>>(w, weight, style_scale,
                                                 fc_weight, fc_bias, wt_bf);
    modconv_kernel<<<dim3(HH, BATCH), 256, 0, stream>>>(x, wt_bf, (float*)d_out);
}
